// MLPDecoder_33586644255339
// MI455X (gfx1250) — matrix operand order repaired and checked by reading, not compiled, not run on hardware
//
#include <hip/hip_runtime.h>
#include <hip/hip_bf16.h>

// ---------------------------------------------------------------------------
// MLPDecoder (NRI decoder) for MI455X / gfx1250.
// Strategy: bf16 WMMA 16x16x32 with f32 accumulation.
//   prep:   fp32 -> bf16 transpose of activations and weights (workspace)
//   edge:   1 workgroup per (b,t,sender n): 32x128 edge_rep tile -> per-k
//           [128x256]+[256x256] WMMA GEMMs, rel-weighted accumulate,
//           column-reduce 31 edges -> agg row, store into AUG buffer
//   node:   1 workgroup per (b,t): 32x320 -> 256 -> 256 -> 64 WMMA MLP,
//           residual add, write output (B,N,T-1,D)
// Loop nests are ordered so each WMMA A tile and B tile is loaded exactly
// once per wave (B tiles shared across the two m-tiles in registers).
// ---------------------------------------------------------------------------

typedef __attribute__((ext_vector_type(16))) __bf16 v16bf;
typedef __attribute__((ext_vector_type(8)))  __bf16 v8bf;
typedef __attribute__((ext_vector_type(8)))  float  v8f;

#define B_ 8
#define N_ 32
#define T_ 64
#define D_ 64
#define K_ 4
#define E_ 992   // N*(N-1)

__device__ __forceinline__ v8f wmma_bf16(v16bf a, v16bf b, v8f c) {
  // (neg_a, A, neg_b, B, c_mod, C, reuse_a, reuse_b)
  return __builtin_amdgcn_wmma_f32_16x16x32_bf16(false, a, false, b, (short)0, c,
                                                 false, false);
}

__device__ __forceinline__ __bf16 to_bf16(float f) { return (__bf16)f; }

__device__ __forceinline__ v8f vzero8() {
  v8f z = {0.f, 0.f, 0.f, 0.f, 0.f, 0.f, 0.f, 0.f};
  return z;
}

// A-matrix tile (16x32 bf16) from row-major bf16 buffer (LDS), ISA layout:
// lane<16: row=l16, K {kt*32+0..7, kt*32+16..23}; lane>=16: row=l16, K {+8..15, +24..31}
__device__ __forceinline__ v16bf load_a(const __bf16* base, int stride, int mt,
                                        int kt, int l16, int half) {
  int row = mt * 16 + l16;
  int kb  = kt * 32 + (half ? 8 : 0);
  const __bf16* p = base + row * stride + kb;
  v8bf lo = *(const v8bf*)p;
  v8bf hi = *(const v8bf*)(p + 16);
  return __builtin_shufflevector(lo, hi, 0, 1, 2, 3, 4, 5, 6, 7,
                                 8, 9, 10, 11, 12, 13, 14, 15);
}

// B-matrix tile (32x16 bf16) from a TRANSPOSED weight (O x I row-major bf16):
// lane<16: col=l16, K {kt*32+0..7, kt*32+16..23}; lane>=16: col=l16, K {+8..15, +24..31}
__device__ __forceinline__ v16bf load_b(const __bf16* wT, int I, int nt, int kt,
                                        int l16, int half) {
  int row = nt * 16 + l16;
  int cb  = kt * 32 + (half ? 8 : 0);
  const __bf16* p = wT + row * I + cb;
  v8bf lo = *(const v8bf*)p;
  v8bf hi = *(const v8bf*)(p + 16);
  return __builtin_shufflevector(lo, hi, 0, 1, 2, 3, 4, 5, 6, 7,
                                 8, 9, 10, 11, 12, 13, 14, 15);
}

// ---------------------------------------------------------------------------
// Prep: inputs (B,N,T,D) f32 -> X (B,T,N,D) bf16 and AUG[...,0:64] bf16
// ---------------------------------------------------------------------------
__global__ __launch_bounds__(256) void prep_x_kernel(
    const float* __restrict__ inputs, __bf16* __restrict__ X,
    __bf16* __restrict__ AUG) {
  int idx = blockIdx.x * 256 + threadIdx.x;  // enumerates (b,t,n,d)
  int d = idx & 63;
  int n = (idx >> 6) & 31;
  int t = (idx >> 11) & 63;
  int b = idx >> 17;
  float v = inputs[(((b * N_ + n) * T_ + t) << 6) + d];
  __bf16 h = to_bf16(v);
  X[idx] = h;
  AUG[(idx >> 6) * 320 + d] = h;
}

// Prep: weight (batch, I, O) f32 -> (batch, O, I) bf16 (transpose + convert)
__global__ __launch_bounds__(256) void prep_w_kernel(
    const float* __restrict__ src, __bf16* __restrict__ dst, int I, int O,
    int total) {
  int idx = blockIdx.x * 256 + threadIdx.x;
  if (idx >= total) return;
  int per = I * O;
  int bb = idx / per, r = idx % per;
  int i = r / O, o = r % O;
  dst[bb * per + o * I + i] = to_bf16(src[idx]);
}

// ---------------------------------------------------------------------------
// Edge MLP: one workgroup per (b, t, sender n). 8 waves, each owns a
// 32-row x 32-col slab (2 m-tiles x 2 n-tiles) of the 32x256 activations.
// ---------------------------------------------------------------------------
__global__ __launch_bounds__(256) void edge_mlp_kernel(
    const __bf16* __restrict__ X, const float* __restrict__ rel_type,
    const __bf16* __restrict__ W1T, const float* __restrict__ b1,
    const __bf16* __restrict__ W2T, const float* __restrict__ b2,
    __bf16* __restrict__ AUG) {
  __shared__ __bf16 sE[32 * 128];   // edge_rep tile (row m = edge (n, j_m))
  __shared__ __bf16 sH[32 * 256];   // hidden h1 (relu'd, bf16)
  __shared__ float  sRel[32 * 4];   // rel_type per row per k (pad row = 0)

  int bid = blockIdx.x;
  int n = bid & 31;
  int t = (bid >> 5) & 63;
  int b = bid >> 11;
  int tid = threadIdx.x;

  // Build edge_rep: row m (<31): [ x[j], x[n] ], j = m<n ? m : m+1. Row 31 pad.
  {
    int row = tid >> 3, seg = tid & 7;
    int col0 = seg * 16;
    int j = (row < 31) ? ((row < n) ? row : row + 1) : n;
    int src = (col0 < 64) ? j : n;
    int sc  = (col0 < 64) ? col0 : (col0 - 64);
    const uint4* p = (const uint4*)(X + (((b * T_ + t) * N_ + src) << 6) + sc);
    uint4* q = (uint4*)(sE + row * 128 + col0);
    q[0] = p[0];
    q[1] = p[1];
  }
  if (tid < 128) {
    int m = tid >> 2, kk = tid & 3;
    sRel[tid] = (m < 31) ? rel_type[(b * E_ + n * 31 + m) * K_ + kk] : 0.0f;
  }
  __syncthreads();

  int w = tid >> 5;        // wave id (wave32)
  int lane = tid & 31;
  int l16 = lane & 15;
  int half = lane >> 4;

  v8f fused[4];            // [mt*2 + ntl]
#pragma unroll
  for (int i = 0; i < 4; i++) fused[i] = vzero8();

  for (int k = 0; k < K_; k++) {
    const __bf16* W1k = W1T + k * 256 * 128;   // (256 out x 128 in)
    const __bf16* W2k = W2T + k * 256 * 256;   // (256 out x 256 in)
    const float* b1k = b1 + k * 256;
    const float* b2k = b2 + k * 256;

    // GEMM1: 32x128 @ 128x256 -> relu -> sH. Each A/B tile loaded once.
    {
      v8f acc[4];
#pragma unroll
      for (int i = 0; i < 4; i++) acc[i] = vzero8();
#pragma unroll
      for (int kt = 0; kt < 4; kt++) {
        v16bf a0 = load_a(sE, 128, 0, kt, l16, half);
        v16bf a1 = load_a(sE, 128, 1, kt, l16, half);
#pragma unroll
        for (int ntl = 0; ntl < 2; ntl++) {
          v16bf bm = load_b(W1k, 128, w * 2 + ntl, kt, l16, half);
          acc[ntl]     = wmma_bf16(a0, bm, acc[ntl]);
          acc[2 + ntl] = wmma_bf16(a1, bm, acc[2 + ntl]);
        }
      }
#pragma unroll
      for (int mt = 0; mt < 2; mt++) {
#pragma unroll
        for (int ntl = 0; ntl < 2; ntl++) {
          int nt = w * 2 + ntl;
          float bias = b1k[nt * 16 + l16];
#pragma unroll
          for (int r = 0; r < 8; r++) {
            float v = acc[mt * 2 + ntl][r] + bias;
            v = v > 0.f ? v : 0.f;
            sH[(mt * 16 + r + half * 8) * 256 + nt * 16 + l16] = to_bf16(v);
          }
        }
      }
    }
    __syncthreads();

    // GEMM2: 32x256 @ 256x256 -> relu -> * rel[row,k] -> fused accumulate
    {
      v8f acc[4];
#pragma unroll
      for (int i = 0; i < 4; i++) acc[i] = vzero8();
#pragma unroll
      for (int kt = 0; kt < 8; kt++) {
        v16bf a0 = load_a(sH, 256, 0, kt, l16, half);
        v16bf a1 = load_a(sH, 256, 1, kt, l16, half);
#pragma unroll
        for (int ntl = 0; ntl < 2; ntl++) {
          v16bf bm = load_b(W2k, 256, w * 2 + ntl, kt, l16, half);
          acc[ntl]     = wmma_bf16(a0, bm, acc[ntl]);
          acc[2 + ntl] = wmma_bf16(a1, bm, acc[2 + ntl]);
        }
      }
#pragma unroll
      for (int mt = 0; mt < 2; mt++) {
#pragma unroll
        for (int ntl = 0; ntl < 2; ntl++) {
          int nt = w * 2 + ntl;
          float bias = b2k[nt * 16 + l16];
#pragma unroll
          for (int r = 0; r < 8; r++) {
            int row = mt * 16 + r + half * 8;
            float v = acc[mt * 2 + ntl][r] + bias;
            v = v > 0.f ? v : 0.f;
            fused[mt * 2 + ntl][r] += v * sRel[row * 4 + k];
          }
        }
      }
    }
    __syncthreads();  // protect sH before next k overwrites it
  }

  // agg[n, col] = sum over 32 rows (pad row contributes 0) of fused
#pragma unroll
  for (int ntl = 0; ntl < 2; ntl++) {
    float s = 0.f;
#pragma unroll
    for (int r = 0; r < 8; r++) s += fused[ntl][r] + fused[2 + ntl][r];
    s += __shfl_xor(s, 16, 32);  // combine the two lane-half row groups
    if (half == 0) {
      int col = w * 32 + ntl * 16 + l16;
      AUG[((b * T_ + t) * N_ + n) * 320 + 64 + col] = to_bf16(s);
    }
  }
}

// ---------------------------------------------------------------------------
// Node MLP: one workgroup per (b, t). aug(32x320) -> 256 -> 256 -> 64 + x.
// ---------------------------------------------------------------------------
__global__ __launch_bounds__(256) void node_mlp_kernel(
    const __bf16* __restrict__ AUG, const float* __restrict__ inputs,
    const __bf16* __restrict__ Wf1T, const float* __restrict__ bf1,
    const __bf16* __restrict__ Wf2T, const float* __restrict__ bf2,
    const __bf16* __restrict__ Wf3T, const float* __restrict__ bf3,
    float* __restrict__ out) {
  __shared__ __bf16 sA[32 * 320];
  __shared__ __bf16 sH1[32 * 256];
  __shared__ __bf16 sH2[32 * 256];

  int bid = blockIdx.x;
  int t = bid & 63;
  int b = bid >> 6;
  int tid = threadIdx.x;

  {  // stage aug tile into LDS: each thread copies 40 bf16 (5 x b128)
    int row = tid >> 3, seg = tid & 7;
    const uint4* p = (const uint4*)(AUG + ((b * T_ + t) * N_ + row) * 320 + seg * 40);
    uint4* q = (uint4*)(sA + row * 320 + seg * 40);
#pragma unroll
    for (int i = 0; i < 5; i++) q[i] = p[i];
  }
  __syncthreads();

  int w = tid >> 5, lane = tid & 31, l16 = lane & 15, half = lane >> 4;

  // GEMM1: 32x320 @ 320x256 -> relu -> sH1
  {
    v8f acc[4];
#pragma unroll
    for (int i = 0; i < 4; i++) acc[i] = vzero8();
#pragma unroll
    for (int kt = 0; kt < 10; kt++) {
      v16bf a0 = load_a(sA, 320, 0, kt, l16, half);
      v16bf a1 = load_a(sA, 320, 1, kt, l16, half);
#pragma unroll
      for (int ntl = 0; ntl < 2; ntl++) {
        v16bf bm = load_b(Wf1T, 320, w * 2 + ntl, kt, l16, half);
        acc[ntl]     = wmma_bf16(a0, bm, acc[ntl]);
        acc[2 + ntl] = wmma_bf16(a1, bm, acc[2 + ntl]);
      }
    }
#pragma unroll
    for (int mt = 0; mt < 2; mt++) {
#pragma unroll
      for (int ntl = 0; ntl < 2; ntl++) {
        int nt = w * 2 + ntl;
        float bias = bf1[nt * 16 + l16];
#pragma unroll
        for (int r = 0; r < 8; r++) {
          float v = acc[mt * 2 + ntl][r] + bias;
          v = v > 0.f ? v : 0.f;
          sH1[(mt * 16 + r + half * 8) * 256 + nt * 16 + l16] = to_bf16(v);
        }
      }
    }
  }
  __syncthreads();

  // GEMM2: 32x256 @ 256x256 -> relu -> sH2
  {
    v8f acc[4];
#pragma unroll
    for (int i = 0; i < 4; i++) acc[i] = vzero8();
#pragma unroll
    for (int kt = 0; kt < 8; kt++) {
      v16bf a0 = load_a(sH1, 256, 0, kt, l16, half);
      v16bf a1 = load_a(sH1, 256, 1, kt, l16, half);
#pragma unroll
      for (int ntl = 0; ntl < 2; ntl++) {
        v16bf bm = load_b(Wf2T, 256, w * 2 + ntl, kt, l16, half);
        acc[ntl]     = wmma_bf16(a0, bm, acc[ntl]);
        acc[2 + ntl] = wmma_bf16(a1, bm, acc[2 + ntl]);
      }
    }
#pragma unroll
    for (int mt = 0; mt < 2; mt++) {
#pragma unroll
      for (int ntl = 0; ntl < 2; ntl++) {
        int nt = w * 2 + ntl;
        float bias = bf2[nt * 16 + l16];
#pragma unroll
        for (int r = 0; r < 8; r++) {
          float v = acc[mt * 2 + ntl][r] + bias;
          v = v > 0.f ? v : 0.f;
          sH2[(mt * 16 + r + half * 8) * 256 + nt * 16 + l16] = to_bf16(v);
        }
      }
    }
  }
  __syncthreads();

  // GEMM3: 32x256 @ 256x64 -> 8 tiles, one per wave; residual add + store
  {
    int mt = w >> 2, nt = w & 3;
    v8f acc = vzero8();
#pragma unroll
    for (int kt = 0; kt < 8; kt++)
      acc = wmma_bf16(load_a(sH2, 256, mt, kt, l16, half),
                      load_b(Wf3T, 256, nt, kt, l16, half), acc);
    float bias = bf3[nt * 16 + l16];
    if (t < T_ - 1) {  // output keeps time steps 0..62
#pragma unroll
      for (int r = 0; r < 8; r++) {
        int node = mt * 16 + r + half * 8;
        int col = nt * 16 + l16;
        float x = inputs[(((b * N_ + node) * T_ + t) << 6) + col];
        out[((b * N_ + node) * (T_ - 1) + t) * 64 + col] = x + acc[r] + bias;
      }
    }
  }
}

// ---------------------------------------------------------------------------
extern "C" void kernel_launch(void* const* d_in, const int* in_sizes, int n_in,
                              void* d_out, int out_size, void* d_ws,
                              size_t ws_size, hipStream_t stream) {
  (void)in_sizes; (void)n_in; (void)out_size; (void)ws_size;
  const float* inputs   = (const float*)d_in[0];
  const float* rel_type = (const float*)d_in[1];
  const float* W1  = (const float*)d_in[4];
  const float* b1  = (const float*)d_in[5];
  const float* W2  = (const float*)d_in[6];
  const float* b2  = (const float*)d_in[7];
  const float* Wf1 = (const float*)d_in[8];
  const float* bf1 = (const float*)d_in[9];
  const float* Wf2 = (const float*)d_in[10];
  const float* bf2 = (const float*)d_in[11];
  const float* Wf3 = (const float*)d_in[12];
  const float* bf3 = (const float*)d_in[13];

  char* ws = (char*)d_ws;
  __bf16* X    = (__bf16*)(ws);                 // (B,T,N,64)   2,097,152 B
  __bf16* AUG  = (__bf16*)(ws + 2097152);       // (B,T,N,320) 10,485,760 B
  __bf16* W1T  = (__bf16*)(ws + 12582912);      // (K,256,128)    262,144 B
  __bf16* W2T  = (__bf16*)(ws + 12845056);      // (K,256,256)    524,288 B
  __bf16* Wf1T = (__bf16*)(ws + 13369344);      // (256,320)      163,840 B
  __bf16* Wf2T = (__bf16*)(ws + 13533184);      // (256,256)      131,072 B
  __bf16* Wf3T = (__bf16*)(ws + 13664256);      // (64,256)        32,768 B

  prep_x_kernel<<<(B_ * T_ * N_ * D_) / 256, 256, 0, stream>>>(inputs, X, AUG);
  prep_w_kernel<<<(K_ * 128 * 256 + 255) / 256, 256, 0, stream>>>(W1, W1T, 128, 256, K_ * 128 * 256);
  prep_w_kernel<<<(K_ * 256 * 256 + 255) / 256, 256, 0, stream>>>(W2, W2T, 256, 256, K_ * 256 * 256);
  prep_w_kernel<<<(320 * 256 + 255) / 256, 256, 0, stream>>>(Wf1, Wf1T, 320, 256, 320 * 256);
  prep_w_kernel<<<(256 * 256 + 255) / 256, 256, 0, stream>>>(Wf2, Wf2T, 256, 256, 256 * 256);
  prep_w_kernel<<<(256 * 64 + 255) / 256, 256, 0, stream>>>(Wf3, Wf3T, 256, 64, 256 * 64);

  edge_mlp_kernel<<<B_ * T_ * N_, 256, 0, stream>>>(X, rel_type, W1T, b1, W2T, b2, AUG);
  node_mlp_kernel<<<B_ * T_, 256, 0, stream>>>(AUG, inputs, Wf1T, bf1, Wf2T, bf2,
                                               Wf3T, bf3, (float*)d_out);
}
